// GNN_Bayes_9234179687244
// MI455X (gfx1250) — matrix operand order repaired and checked by reading, not compiled, not run on hardware
//
#include <hip/hip_runtime.h>
#include <hip/hip_bf16.h>

// ---------------------------------------------------------------------------
// Problem constants (match reference)
// ---------------------------------------------------------------------------
#define NN 50000
#define EE 500000
#define GG 2048
#define IN_ATOM 128
#define HH 256
#define OUTW 2

typedef __attribute__((ext_vector_type(16))) __bf16       v16bf;
typedef __attribute__((ext_vector_type(8)))  float        v8f;
typedef __attribute__((ext_vector_type(4)))  float        v4f;
typedef __attribute__((ext_vector_type(8)))  unsigned int v8u;
typedef __attribute__((ext_vector_type(4)))  unsigned int v4u;

// fp32 -> bf16 (round to nearest even), packed helpers
__device__ __forceinline__ unsigned short f2bf(float f) {
    unsigned int u = __builtin_bit_cast(unsigned int, f);
    unsigned int r = u + 0x7FFFu + ((u >> 16) & 1u);
    return (unsigned short)(r >> 16);
}
__device__ __forceinline__ unsigned int pack2(float a, float b) {
    return (unsigned int)f2bf(a) | ((unsigned int)f2bf(b) << 16);
}

// ---------------------------------------------------------------------------
// Utility kernels
// ---------------------------------------------------------------------------
__global__ void __launch_bounds__(256) zero_kernel(float* __restrict__ p, int n) {
    int i = blockIdx.x * 256 + threadIdx.x;
    if (i < n) p[i] = 0.0f;
}

// dst[c*K + k] = bf16(src[k*Nc + c])   (transpose + convert)
__global__ void __launch_bounds__(256) wtrans_kernel(const float* __restrict__ W,
                                                     unsigned short* __restrict__ Wt,
                                                     int K, int Nc) {
    int idx = blockIdx.x * 256 + threadIdx.x;
    if (idx >= K * Nc) return;
    int k = idx / Nc, c = idx - k * Nc;
    Wt[c * K + k] = f2bf(W[idx]);
}

// Bayesian head W1: Wt[c*256 + k] = bf16(mu + exp(ls) * eps) (transposed)
__global__ void __launch_bounds__(256) headw_kernel(const float* __restrict__ mu,
                                                    const float* __restrict__ ls,
                                                    const float* __restrict__ ep,
                                                    unsigned short* __restrict__ Wt) {
    int idx = blockIdx.x * 256 + threadIdx.x;
    if (idx >= HH * HH) return;
    int k = idx >> 8, c = idx & 255;
    float v = mu[idx] + __expf(ls[idx]) * ep[idx];
    Wt[c * HH + k] = f2bf(v);
}

__global__ void __launch_bounds__(256) headb_kernel(const float* __restrict__ mu,
                                                    const float* __restrict__ ls,
                                                    const float* __restrict__ ep,
                                                    float* __restrict__ b, int n) {
    int i = blockIdx.x * 256 + threadIdx.x;
    if (i < n) b[i] = mu[i] + __expf(ls[i]) * ep[i];
}

// ---------------------------------------------------------------------------
// Edge projection + message + scatter-add (VALU; K=16 too small for WMMA).
// One wave per edge, lane j handles features j, j+32, ...
// edge_attr rows staged in LDS; h[src] row prefetched (global_prefetch_b8).
// ---------------------------------------------------------------------------
__global__ void __launch_bounds__(256)
edge_msg_kernel(const float* __restrict__ ea, const int* __restrict__ ei,
                const float* __restrict__ Weg, const float* __restrict__ beg,
                const float* __restrict__ h, float* __restrict__ agg, int D, int E) {
    __shared__ float WeS[16 * HH];
    __shared__ float beS[HH];
    __shared__ float eaS[8 * 16];
    const int tid = threadIdx.x;
    const int eBase = blockIdx.x * 8;
    for (int i = tid; i < 16 * D; i += 256) WeS[i] = Weg[i];
    for (int i = tid; i < D; i += 256) beS[i] = beg[i];
    if (tid < 128) {
        int e = eBase + (tid >> 4);
        eaS[tid] = (e < E) ? ea[e * 16 + (tid & 15)] : 0.0f;
    }
    __syncthreads();

    int wv = tid >> 5, lane = tid & 31;
    int e = eBase + wv;
    if (e >= E) return;
    int src = ei[e], dst = ei[E + e];
    __builtin_prefetch(&h[src * D], 0, 0);

    float ea16[16];
#pragma unroll
    for (int k = 0; k < 16; k++) ea16[k] = eaS[wv * 16 + k];

    for (int i = 0; i < D / 32; i++) {
        int j = lane + 32 * i;
        float acc = beS[j];
#pragma unroll
        for (int k = 0; k < 16; k++) acc += ea16[k] * WeS[k * D + j];
        float m = h[src * D + j] + acc;
        m = m > 0.0f ? m : 0.0f;
        atomicAdd(&agg[dst * D + j], m);
    }
}

// ---------------------------------------------------------------------------
// WMMA fragment loaders
// A fragment (16x32 bf16): lane holds row (lane&15), K = kb..kb+7, kb+16..kb+23,
//   kb = 0 (lanes 0-15) or 8 (lanes 16-31).  Two 16B contiguous chunks.
// B fragment (32x16 bf16): lane holds col (lane&15), K = kb..kb+7, kb+16..kb+23,
//   kb = 0 (lanes 0-15) or 8 (lanes 16-31).  Two 16B contiguous chunks.
// ---------------------------------------------------------------------------
__device__ __forceinline__ v16bf load_a_frag_lds(const unsigned int* zrow, int kbase, int kbA) {
    v4u lo = *(const v4u*)&zrow[(kbase + kbA) >> 1];
    v4u hi = *(const v4u*)&zrow[(kbase + kbA + 16) >> 1];
    v8u a;
    a[0] = lo[0]; a[1] = lo[1]; a[2] = lo[2]; a[3] = lo[3];
    a[4] = hi[0]; a[5] = hi[1]; a[6] = hi[2]; a[7] = hi[3];
    return __builtin_bit_cast(v16bf, a);
}

// A fragment from bf16 (ushort) LDS row: consecutive bf16 = packed pairs already
__device__ __forceinline__ v16bf load_a_frag_bf(const unsigned short* zrow, int kbase, int kbA) {
    v4u lo = *(const v4u*)&zrow[kbase + kbA];
    v4u hi = *(const v4u*)&zrow[kbase + kbA + 16];
    v8u a;
    a[0] = lo[0]; a[1] = lo[1]; a[2] = lo[2]; a[3] = lo[3];
    a[4] = hi[0]; a[5] = hi[1]; a[6] = hi[2]; a[7] = hi[3];
    return __builtin_bit_cast(v16bf, a);
}

__device__ __forceinline__ v16bf load_b_frag(const unsigned short* wp) {
    v4u lo = *(const v4u*)wp;
    v4u hi = *(const v4u*)(wp + 16);
    v8u b;
    b[0] = lo[0]; b[1] = lo[1]; b[2] = lo[2]; b[3] = lo[3];
    b[4] = hi[0]; b[5] = hi[1]; b[6] = hi[2]; b[7] = hi[3];
    return __builtin_bit_cast(v16bf, b);
}

// ---------------------------------------------------------------------------
// Fused GINE node MLP:  h_out = relu( relu( z @ W1 + b1 ) @ W2 + b2 )
// with z = (1+eps)*h_in + agg   (built in LDS as packed bf16)
// 64 nodes/block, 256 threads = 8 waves = 2 row-halves (32 rows = 2 M-tiles
// per wave, sharing each B fragment) x 4 N-quarters.
// W1t: bf16 [256][D] (transposed), W2t: bf16 [256][256] (transposed)
// LDS: zA (bf16 pairs) + z1 (bf16) = 64KB max.
// ---------------------------------------------------------------------------
template <int D>
__global__ void __launch_bounds__(256)
mlp_kernel(const float* __restrict__ hin, const float* __restrict__ agg,
           const float* __restrict__ epsp,
           const unsigned short* __restrict__ W1t, const float* __restrict__ b1,
           const unsigned short* __restrict__ W2t, const float* __restrict__ b2,
           float* __restrict__ hout, int Nn) {
    __shared__ unsigned int   zA[64 * (D / 2)];   // packed bf16 pairs of z
    __shared__ unsigned short z1[64 * HH];        // bf16 intermediate

    const int tid = threadIdx.x;
    const int lane = tid & 31;
    const int wv = tid >> 5;
    const int wm = wv & 1;       // row half: 32 rows = 2 WMMA M-tiles
    const int nq = wv >> 1;      // N quarter (64 cols = 4 N-tiles)
    const int nodeBase = blockIdx.x * 64;
    const float epsv = 1.0f + epsp[0];

    // Build z = (1+eps)*h + agg as packed bf16 in LDS (vectorized, 16B stores)
    for (int idx = tid; idx < 64 * (D / 8); idx += 256) {
        int r = idx / (D / 8);
        int q = idx - r * (D / 8);
        int n = nodeBase + r;
        v4u w;
        if (n < Nn) {
            const float* hp = &hin[n * D + q * 8];
            const float* gp = &agg[n * D + q * 8];
            v4f h0 = *(const v4f*)hp;
            v4f h1 = *(const v4f*)(hp + 4);
            v4f g0 = *(const v4f*)gp;
            v4f g1 = *(const v4f*)(gp + 4);
            w[0] = pack2(epsv * h0[0] + g0[0], epsv * h0[1] + g0[1]);
            w[1] = pack2(epsv * h0[2] + g0[2], epsv * h0[3] + g0[3]);
            w[2] = pack2(epsv * h1[0] + g1[0], epsv * h1[1] + g1[1]);
            w[3] = pack2(epsv * h1[2] + g1[2], epsv * h1[3] + g1[3]);
        } else {
            w[0] = 0u; w[1] = 0u; w[2] = 0u; w[3] = 0u;
        }
        *(v4u*)&zA[r * (D / 2) + q * 4] = w;
    }
    __syncthreads();

    const int rA0  = wm * 32 + (lane & 15);       // first M-tile row
    const int kbA  = (lane < 16) ? 0 : 8;
    const int colB = nq * 64 + (lane & 15);
    const int kbB  = (lane < 16) ? 0 : 8;
    const int rowBase = (lane < 16) ? 0 : 8;

    // ----- stage 1: z1 = bf16(relu(z @ W1 + b1)) -----
    {
        v8f acc0[4] = {};
        v8f acc1[4] = {};
#pragma unroll
        for (int ks = 0; ks < D / 32; ks++) {
            v16bf a0 = load_a_frag_lds(&zA[rA0 * (D / 2)], ks * 32, kbA);
            v16bf a1 = load_a_frag_lds(&zA[(rA0 + 16) * (D / 2)], ks * 32, kbA);
#pragma unroll
            for (int nt = 0; nt < 4; nt++) {
                v16bf bv = load_b_frag(&W1t[(colB + nt * 16) * D + ks * 32 + kbB]);
                acc0[nt] = __builtin_amdgcn_wmma_f32_16x16x32_bf16(
                    false, a0, false, bv, (short)0, acc0[nt], false, false);
                acc1[nt] = __builtin_amdgcn_wmma_f32_16x16x32_bf16(
                    false, a1, false, bv, (short)0, acc1[nt], false, false);
            }
        }
#pragma unroll
        for (int nt = 0; nt < 4; nt++) {
            int c = nq * 64 + nt * 16 + (lane & 15);
            float bb = b1[c];
#pragma unroll
            for (int i = 0; i < 8; i++) {
                float v0 = acc0[nt][i] + bb;
                float v1 = acc1[nt][i] + bb;
                z1[(wm * 32 + rowBase + i) * HH + c]      = f2bf(v0 > 0.0f ? v0 : 0.0f);
                z1[(wm * 32 + 16 + rowBase + i) * HH + c] = f2bf(v1 > 0.0f ? v1 : 0.0f);
            }
        }
    }
    __syncthreads();

    // ----- stage 2: h_out = relu(z1 @ W2 + b2) -----
    {
        v8f acc0[4] = {};
        v8f acc1[4] = {};
#pragma unroll
        for (int ks = 0; ks < HH / 32; ks++) {
            v16bf a0 = load_a_frag_bf(&z1[rA0 * HH], ks * 32, kbA);
            v16bf a1 = load_a_frag_bf(&z1[(rA0 + 16) * HH], ks * 32, kbA);
#pragma unroll
            for (int nt = 0; nt < 4; nt++) {
                v16bf bv = load_b_frag(&W2t[(colB + nt * 16) * HH + ks * 32 + kbB]);
                acc0[nt] = __builtin_amdgcn_wmma_f32_16x16x32_bf16(
                    false, a0, false, bv, (short)0, acc0[nt], false, false);
                acc1[nt] = __builtin_amdgcn_wmma_f32_16x16x32_bf16(
                    false, a1, false, bv, (short)0, acc1[nt], false, false);
            }
        }
#pragma unroll
        for (int nt = 0; nt < 4; nt++) {
            int c = nq * 64 + nt * 16 + (lane & 15);
            float bb = b2[c];
#pragma unroll
            for (int i = 0; i < 8; i++) {
                int n0 = nodeBase + wm * 32 + rowBase + i;
                int n1 = n0 + 16;
                if (n0 < Nn) {
                    float v = acc0[nt][i] + bb;
                    hout[n0 * HH + c] = v > 0.0f ? v : 0.0f;
                }
                if (n1 < Nn) {
                    float v = acc1[nt][i] + bb;
                    hout[n1 * HH + c] = v > 0.0f ? v : 0.0f;
                }
            }
        }
    }
}

// ---------------------------------------------------------------------------
// global_add_pool: hg[batch[n]] += h[n]
// ---------------------------------------------------------------------------
__global__ void __launch_bounds__(256)
pool_kernel(const float* __restrict__ h, const int* __restrict__ batch,
            float* __restrict__ hg, int Nn) {
    int idx = blockIdx.x * 256 + threadIdx.x;
    if (idx >= Nn * HH) return;
    int n = idx >> 8, c = idx & 255;
    atomicAdd(&hg[batch[n] * HH + c], h[idx]);
}

// ---------------------------------------------------------------------------
// Head stage 1: h1 = silu(hg @ w1eff + b1eff), WMMA (K = 256, N = 256)
// ---------------------------------------------------------------------------
__global__ void __launch_bounds__(256)
head_gemm_silu(const float* __restrict__ A, const unsigned short* __restrict__ Wt,
               const float* __restrict__ bias, float* __restrict__ out, int M) {
    __shared__ unsigned int sA[32 * (HH / 2)];

    const int tid = threadIdx.x;
    const int lane = tid & 31;
    const int wv = tid >> 5;
    const int mt = wv & 1;
    const int nq = wv >> 1;
    const int rowBlk = blockIdx.x * 32;

    for (int idx = tid; idx < 32 * (HH / 2); idx += 256) {
        int r = idx / (HH / 2);
        int w = idx - r * (HH / 2);
        int m = rowBlk + r;
        float a0 = 0.0f, a1 = 0.0f;
        if (m < M) {
            a0 = A[m * HH + 2 * w];
            a1 = A[m * HH + 2 * w + 1];
        }
        sA[idx] = pack2(a0, a1);
    }
    __syncthreads();

    const int rA  = mt * 16 + (lane & 15);
    const int kbA = (lane < 16) ? 0 : 8;
    const int colB = nq * 64 + (lane & 15);
    const int kbB = (lane < 16) ? 0 : 8;
    const int rowBase = mt * 16 + ((lane < 16) ? 0 : 8);

    v8f acc[4] = {};
#pragma unroll
    for (int ks = 0; ks < HH / 32; ks++) {
        v16bf av = load_a_frag_lds(&sA[rA * (HH / 2)], ks * 32, kbA);
#pragma unroll
        for (int nt = 0; nt < 4; nt++) {
            int c = colB + nt * 16;
            v16bf bv = load_b_frag(&Wt[c * HH + ks * 32 + kbB]);
            acc[nt] = __builtin_amdgcn_wmma_f32_16x16x32_bf16(
                false, av, false, bv, (short)0, acc[nt], false, false);
        }
    }
#pragma unroll
    for (int nt = 0; nt < 4; nt++) {
        int c = nq * 64 + nt * 16 + (lane & 15);
        float bb = bias[c];
#pragma unroll
        for (int i = 0; i < 8; i++) {
            int m = rowBlk + rowBase + i;
            if (m < M) {
                float v = acc[nt][i] + bb;
                out[m * HH + c] = v / (1.0f + __expf(-v));   // silu
            }
        }
    }
}

// ---------------------------------------------------------------------------
// Head stage 2: out[g, 0..1] = h1[g,:] @ w2eff + b2eff  (N=2, VALU)
// ---------------------------------------------------------------------------
__global__ void __launch_bounds__(256)
headout_kernel(const float* __restrict__ h1,
               const float* __restrict__ wmu, const float* __restrict__ wls,
               const float* __restrict__ wep,
               const float* __restrict__ bmu, const float* __restrict__ bls,
               const float* __restrict__ bep, float* __restrict__ out) {
    __shared__ float w2[HH * OUTW];
    __shared__ float b2s[OUTW];
    for (int i = threadIdx.x; i < HH * OUTW; i += 256)
        w2[i] = wmu[i] + __expf(wls[i]) * wep[i];
    if (threadIdx.x < OUTW)
        b2s[threadIdx.x] = bmu[threadIdx.x] + __expf(bls[threadIdx.x]) * bep[threadIdx.x];
    __syncthreads();

    int g = blockIdx.x * 256 + threadIdx.x;
    if (g >= GG) return;
    float a0 = b2s[0], a1 = b2s[1];
    for (int k = 0; k < HH; k++) {
        float hv = h1[g * HH + k];
        a0 += hv * w2[k * 2];
        a1 += hv * w2[k * 2 + 1];
    }
    out[g * 2] = a0;
    out[g * 2 + 1] = a1;
}

// ---------------------------------------------------------------------------
// Host-side orchestration
// ---------------------------------------------------------------------------
extern "C" void kernel_launch(void* const* d_in, const int* in_sizes, int n_in,
                              void* d_out, int out_size, void* d_ws, size_t ws_size,
                              hipStream_t stream) {
    (void)in_sizes; (void)n_in; (void)out_size; (void)ws_size;

    const float* x         = (const float*)d_in[0];
    const float* edge_attr = (const float*)d_in[1];
    const int*   edge_idx  = (const int*)d_in[2];
    const int*   batch     = (const int*)d_in[3];

    // per-layer params: base index 4 + 7*layer : We, be, W1, b1, W2, b2, eps
    const float* We[3];  const float* be[3];
    const float* W1[3];  const float* b1[3];
    const float* W2[3];  const float* b2[3];
    const float* eps[3];
    for (int l = 0; l < 3; l++) {
        int b = 4 + 7 * l;
        We[l]  = (const float*)d_in[b + 0];
        be[l]  = (const float*)d_in[b + 1];
        W1[l]  = (const float*)d_in[b + 2];
        b1[l]  = (const float*)d_in[b + 3];
        W2[l]  = (const float*)d_in[b + 4];
        b2[l]  = (const float*)d_in[b + 5];
        eps[l] = (const float*)d_in[b + 6];
    }
    const float* h1_wmu = (const float*)d_in[25];
    const float* h1_wls = (const float*)d_in[26];
    const float* h1_bmu = (const float*)d_in[27];
    const float* h1_bls = (const float*)d_in[28];
    const float* h2_wmu = (const float*)d_in[29];
    const float* h2_wls = (const float*)d_in[30];
    const float* h2_bmu = (const float*)d_in[31];
    const float* h2_bls = (const float*)d_in[32];
    const float* eps_w1 = (const float*)d_in[33];
    const float* eps_b1 = (const float*)d_in[34];
    const float* eps_w2 = (const float*)d_in[35];
    const float* eps_b2 = (const float*)d_in[36];

    // workspace carve-up (256B aligned slabs)
    char* ws = (char*)d_ws;
    size_t off = 0;
    auto alloc = [&](size_t bytes) -> void* {
        void* p = ws + off;
        off += (bytes + 255) & ~(size_t)255;
        return p;
    };
    float* hA  = (float*)alloc((size_t)NN * HH * 4);
    float* hB  = (float*)alloc((size_t)NN * HH * 4);
    float* agg = (float*)alloc((size_t)NN * HH * 4);
    float* hg  = (float*)alloc((size_t)GG * HH * 4);
    float* h1b = (float*)alloc((size_t)GG * HH * 4);
    unsigned short* W1t[3];
    unsigned short* W2t[3];
    W1t[0] = (unsigned short*)alloc((size_t)HH * IN_ATOM * 2);
    W2t[0] = (unsigned short*)alloc((size_t)HH * HH * 2);
    W1t[1] = (unsigned short*)alloc((size_t)HH * HH * 2);
    W2t[1] = (unsigned short*)alloc((size_t)HH * HH * 2);
    W1t[2] = (unsigned short*)alloc((size_t)HH * HH * 2);
    W2t[2] = (unsigned short*)alloc((size_t)HH * HH * 2);
    unsigned short* hw1t = (unsigned short*)alloc((size_t)HH * HH * 2);
    float* b1eff = (float*)alloc((size_t)HH * 4);

    const int dimIn[3] = {IN_ATOM, HH, HH};

    // --- weight prep (bf16, transposed) ---
    for (int l = 0; l < 3; l++) {
        int k1 = dimIn[l];
        wtrans_kernel<<<(k1 * HH + 255) / 256, 256, 0, stream>>>(W1[l], W1t[l], k1, HH);
        wtrans_kernel<<<(HH * HH + 255) / 256, 256, 0, stream>>>(W2[l], W2t[l], HH, HH);
    }
    headw_kernel<<<(HH * HH + 255) / 256, 256, 0, stream>>>(h1_wmu, h1_wls, eps_w1, hw1t);
    headb_kernel<<<1, 256, 0, stream>>>(h1_bmu, h1_bls, eps_b1, b1eff, HH);

    // --- GINE layers ---
    const float* hcur = x;
    float* houts[3] = {hA, hB, hA};
    for (int l = 0; l < 3; l++) {
        int D = dimIn[l];
        zero_kernel<<<((size_t)NN * D + 255) / 256, 256, 0, stream>>>(agg, NN * D);
        edge_msg_kernel<<<(EE + 7) / 8, 256, 0, stream>>>(
            edge_attr, edge_idx, We[l], be[l], hcur, agg, D, EE);
        int blocks = (NN + 63) / 64;
        if (D == IN_ATOM) {
            mlp_kernel<IN_ATOM><<<blocks, 256, 0, stream>>>(
                hcur, agg, eps[l], W1t[l], b1[l], W2t[l], b2[l], houts[l], NN);
        } else {
            mlp_kernel<HH><<<blocks, 256, 0, stream>>>(
                hcur, agg, eps[l], W1t[l], b1[l], W2t[l], b2[l], houts[l], NN);
        }
        hcur = houts[l];
    }

    // --- pooling ---
    zero_kernel<<<(GG * HH + 255) / 256, 256, 0, stream>>>(hg, GG * HH);
    pool_kernel<<<((size_t)NN * HH + 255) / 256, 256, 0, stream>>>(hcur, batch, hg, NN);

    // --- Bayesian head ---
    head_gemm_silu<<<(GG + 31) / 32, 256, 0, stream>>>(hg, hw1t, b1eff, h1b, GG);
    headout_kernel<<<(GG + 255) / 256, 256, 0, stream>>>(
        h1b, h2_wmu, h2_wls, eps_w2, h2_bmu, h2_bls, eps_b2, (float*)d_out);
}
